// GroupedExpertsTE_52707838657314
// MI455X (gfx1250) — matrix operand order repaired and checked by reading, not compiled, not run on hardware
//
#include <hip/hip_runtime.h>
#include <hip/hip_bf16.h>

typedef __bf16 bf16_t;
typedef __attribute__((ext_vector_type(16))) __bf16 v16bf;
typedef __attribute__((ext_vector_type(8)))  float  v8f;

#define N_EXPERTS        8
#define DIM              2048
#define INTER            1408
#define TOK_PER_E        2048
#define TOTAL_TOKENS     16384

#define BM 128
#define BN 128
#define BK 64
#define LDK (BK + 8)   // padded LDS row stride (elements) to break bank conflicts

union FragU { v16bf v; uint4 q[2]; };
union Pack4 { bf16_t h[4]; uint2 u; };

// ---------------------------------------------------------------------------
// CDNA5 async global->LDS copy (GLOBAL_LOAD_ASYNC_TO_LDS_B128, ASYNCcnt).
// GVS addressing: mem = SGPR64 + VGPR32_offset ; LDS byte address comes from the vdst VGPR.
__device__ __forceinline__ void async_copy_b128(unsigned lds_off,
                                                const void* sbase,
                                                unsigned gbyte_off) {
    asm volatile("global_load_async_to_lds_b128 %0, %1, %2"
                 :
                 : "v"(lds_off), "v"(gbyte_off), "s"(sbase)
                 : "memory");
}
__device__ __forceinline__ void wait_async0() {
    asm volatile("s_wait_asynccnt 0x0" ::: "memory");
}
__device__ __forceinline__ unsigned lds_off32(const void* p) {
    // flat LDS addresses carry the wave-relative LDS byte offset in addr[31:0]
    return (unsigned)(unsigned long long)p;
}

// ---------------------------------------------------------------------------
// A fragment from LDS (16-bit A 16x32 per ISA layout):
// lane L: M = L&15, hi = L>>4; elems 0..7 -> K = kb + 8*hi + e,
// elems 8..15 -> K = kb + 16 + 8*hi + (e-8). Two 16B ds loads.
__device__ __forceinline__ v16bf lds_a_frag(const bf16_t* As, int mrow, int lane, int kb) {
    int m  = mrow + (lane & 15);
    int hi = lane >> 4;
    const bf16_t* p = As + m * LDK + kb + 8 * hi;
    FragU u;
    u.q[0] = *(const uint4*)(p);
    u.q[1] = *(const uint4*)(p + 16);
    return u.v;
}

// B fragment from LDS (16-bit B KxN layout, B stored N-major / K-contiguous):
// lane L: N = L&15, hi = L>>4; elems 0..7 -> K = kb + 8*hi + e, elems 8..15 -> K = kb + 16 + 8*hi + (e-8).
__device__ __forceinline__ v16bf lds_b_frag(const bf16_t* Bs, int ncol, int lane, int kb) {
    int n  = ncol + (lane & 15);
    int hi = lane >> 4;
    const bf16_t* p = Bs + n * LDK + kb + 8 * hi;
    FragU u;
    u.q[0] = *(const uint4*)(p);
    u.q[1] = *(const uint4*)(p + 16);
    return u.v;
}

// ---------------------------------------------------------------------------
// fp32 -> bf16 elementwise (x)
__global__ __launch_bounds__(256) void convert_x_kernel(const float* __restrict__ x,
                                                        bf16_t* __restrict__ xb) {
    size_t i = ((size_t)blockIdx.x * 256 + threadIdx.x) * 4;
    float4 f = *(const float4*)(x + i);
    Pack4 r;
    r.h[0] = (bf16_t)f.x; r.h[1] = (bf16_t)f.y;
    r.h[2] = (bf16_t)f.z; r.h[3] = (bf16_t)f.w;
    *(uint2*)(xb + i) = r.u;
}

// Transpose + de-interleave gate_up: [E][K=DIM][2*INTER] f32 ->
// [E][2*INTER][DIM] bf16 with row map f -> (f&1 ? INTER + f/2 : f/2)
__global__ __launch_bounds__(256) void transpose_gup_kernel(const float* __restrict__ gw,
                                                            bf16_t* __restrict__ gupT) {
    __shared__ float t[32][33];
    const int e  = blockIdx.z;
    const int fb = blockIdx.x * 32;   // over 2*INTER
    const int kb = blockIdx.y * 32;   // over DIM
    const int tx = threadIdx.x, ty = threadIdx.y;   // (32,8)
    const float* src = gw + (size_t)e * DIM * (2 * INTER);
#pragma unroll
    for (int j = 0; j < 4; ++j) {
        int k = kb + ty + j * 8;
        t[ty + j * 8][tx] = src[(size_t)k * (2 * INTER) + fb + tx];
    }
    __syncthreads();
    bf16_t* dst = gupT + (size_t)e * (2 * INTER) * DIM;
#pragma unroll
    for (int j = 0; j < 4; ++j) {
        int f = fb + ty + j * 8;
        int n = (f & 1) ? (INTER + (f >> 1)) : (f >> 1);
        dst[(size_t)n * DIM + kb + tx] = (bf16_t)t[tx][ty + j * 8];
    }
}

// Transpose down: [E][K=INTER][DIM] f32 -> [E][DIM][INTER] bf16
__global__ __launch_bounds__(256) void transpose_down_kernel(const float* __restrict__ dw,
                                                             bf16_t* __restrict__ dT) {
    __shared__ float t[32][33];
    const int e  = blockIdx.z;
    const int fb = blockIdx.x * 32;   // over DIM (output rows)
    const int kb = blockIdx.y * 32;   // over INTER
    const int tx = threadIdx.x, ty = threadIdx.y;
    const float* src = dw + (size_t)e * INTER * DIM;
#pragma unroll
    for (int j = 0; j < 4; ++j) {
        int k = kb + ty + j * 8;
        t[ty + j * 8][tx] = src[(size_t)k * DIM + fb + tx];
    }
    __syncthreads();
    bf16_t* dst = dT + (size_t)e * DIM * INTER;
#pragma unroll
    for (int j = 0; j < 4; ++j) {
        int f = fb + ty + j * 8;
        dst[(size_t)f * INTER + kb + tx] = (bf16_t)t[tx][ty + j * 8];
    }
}

// ---------------------------------------------------------------------------
// GEMM1 + GEGLU epilogue: act = geglu(x @ W_gu) * probs   (bf16 out)
// 128x128 block (gate + matching up tile), 8 waves, wave tile 64x32,
// double-buffered LDS fed by async global->LDS copies.
__global__ __launch_bounds__(256) void moe_gemm1_kernel(const bf16_t* __restrict__ Xb,
                                                        const bf16_t* __restrict__ gupT,
                                                        const float* __restrict__ probs,
                                                        bf16_t* __restrict__ act) {
    __shared__ bf16_t As[2][BM * LDK];
    __shared__ bf16_t Bs[2][2][BN * LDK];

    const int e    = blockIdx.z;
    const int m0   = blockIdx.y * BM;
    const int n0   = blockIdx.x * BN;
    const int tid  = threadIdx.x;
    const int lane = tid & 31;
    const int wave = tid >> 5;
    const int wm   = (wave >> 2) * 64;  // 0 / 64
    const int wn   = (wave & 3) * 32;   // 0..96

    const bf16_t* Ag  = Xb   + (size_t)(e * TOK_PER_E + m0) * DIM;
    const bf16_t* Bg0 = gupT + (size_t)e * (2 * INTER) * DIM + (size_t)n0 * DIM;
    const bf16_t* Bg1 = Bg0  + (size_t)INTER * DIM;

    v8f accg[4][2], accu[4][2];
#pragma unroll
    for (int i = 0; i < 4; ++i)
#pragma unroll
        for (int j = 0; j < 2; ++j) { accg[i][j] = (v8f){}; accu[i][j] = (v8f){}; }

    // issue one stage's async copies into LDS buffer `buf`
    auto issue_stage = [&](int buf, int k0) {
#pragma unroll
        for (int i = 0; i < 4; ++i) {
            int idx = tid + i * 256;           // 1024 x b128 per 16KB tile
            int row = idx >> 3;
            int cq  = (idx & 7) * 8;           // element offset in K
            unsigned go = (unsigned)((row * DIM + k0 + cq) * 2);
            async_copy_b128(lds_off32(&As[buf][row * LDK + cq]),    Ag,  go);
            async_copy_b128(lds_off32(&Bs[buf][0][row * LDK + cq]), Bg0, go);
            async_copy_b128(lds_off32(&Bs[buf][1][row * LDK + cq]), Bg1, go);
        }
    };

    constexpr int NSTAGES = DIM / BK;   // 32
    issue_stage(0, 0);
    wait_async0();
    __syncthreads();

    for (int s = 0; s < NSTAGES; ++s) {
        const int cur = s & 1;
        if (s + 1 < NSTAGES) issue_stage(cur ^ 1, (s + 1) * BK);
#pragma unroll
        for (int kb = 0; kb < BK; kb += 32) {
            v16bf af[4];
#pragma unroll
            for (int i = 0; i < 4; ++i) af[i] = lds_a_frag(As[cur], wm + 16 * i, lane, kb);
#pragma unroll
            for (int j = 0; j < 2; ++j) {
                v16bf bg = lds_b_frag(Bs[cur][0], wn + 16 * j, lane, kb);
                v16bf bu = lds_b_frag(Bs[cur][1], wn + 16 * j, lane, kb);
#pragma unroll
                for (int i = 0; i < 4; ++i) {
                    accg[i][j] = __builtin_amdgcn_wmma_f32_16x16x32_bf16(
                        false, af[i], false, bg, (short)0, accg[i][j], false, false);
                    accu[i][j] = __builtin_amdgcn_wmma_f32_16x16x32_bf16(
                        false, af[i], false, bu, (short)0, accu[i][j], false, false);
                }
            }
        }
        wait_async0();       // own next-stage copies landed in LDS
        __syncthreads();     // everyone done reading `cur` / writing `cur^1`
    }

    // GEGLU epilogue: gate=min(g,7); up=clip(u,-7,7); glu=g*sigmoid(1.702*g);
    // act = glu*(up+1)*prob
    const int hi = lane >> 4;
    const int ln = lane & 15;
#pragma unroll
    for (int i = 0; i < 4; ++i) {
#pragma unroll
        for (int j = 0; j < 2; ++j) {
            int n = n0 + wn + 16 * j + ln;
#pragma unroll
            for (int v = 0; v < 8; ++v) {
                int m   = m0 + wm + 16 * i + v + 8 * hi;
                int tok = e * TOK_PER_E + m;
                float g = fminf(accg[i][j][v], 7.0f);
                float u = fminf(fmaxf(accu[i][j][v], -7.0f), 7.0f);
                float glu = g / (1.0f + __expf(-1.702f * g));
                float val = glu * (u + 1.0f) * probs[tok];
                act[(size_t)tok * INTER + n] = (bf16_t)val;
            }
        }
    }
}

// ---------------------------------------------------------------------------
// GEMM2: out = act @ W_dn  (fp32 out), same async double-buffered pipeline.
__global__ __launch_bounds__(256) void moe_gemm2_kernel(const bf16_t* __restrict__ Ab,
                                                        const bf16_t* __restrict__ dT,
                                                        float* __restrict__ out) {
    __shared__ bf16_t As[2][BM * LDK];
    __shared__ bf16_t Bs[2][BN * LDK];

    const int e    = blockIdx.z;
    const int m0   = blockIdx.y * BM;
    const int n0   = blockIdx.x * BN;
    const int tid  = threadIdx.x;
    const int lane = tid & 31;
    const int wave = tid >> 5;
    const int wm   = (wave >> 2) * 64;
    const int wn   = (wave & 3) * 32;

    const bf16_t* Ag = Ab + (size_t)(e * TOK_PER_E + m0) * INTER;
    const bf16_t* Bg = dT + (size_t)e * DIM * INTER + (size_t)n0 * INTER;

    v8f acc[4][2];
#pragma unroll
    for (int i = 0; i < 4; ++i)
#pragma unroll
        for (int j = 0; j < 2; ++j) acc[i][j] = (v8f){};

    auto issue_stage = [&](int buf, int k0) {
#pragma unroll
        for (int i = 0; i < 4; ++i) {
            int idx = tid + i * 256;
            int row = idx >> 3;
            int cq  = (idx & 7) * 8;
            unsigned go = (unsigned)((row * INTER + k0 + cq) * 2);
            async_copy_b128(lds_off32(&As[buf][row * LDK + cq]), Ag, go);
            async_copy_b128(lds_off32(&Bs[buf][row * LDK + cq]), Bg, go);
        }
    };

    constexpr int NSTAGES = INTER / BK;   // 22
    issue_stage(0, 0);
    wait_async0();
    __syncthreads();

    for (int s = 0; s < NSTAGES; ++s) {
        const int cur = s & 1;
        if (s + 1 < NSTAGES) issue_stage(cur ^ 1, (s + 1) * BK);
#pragma unroll
        for (int kb = 0; kb < BK; kb += 32) {
            v16bf af[4];
#pragma unroll
            for (int i = 0; i < 4; ++i) af[i] = lds_a_frag(As[cur], wm + 16 * i, lane, kb);
#pragma unroll
            for (int j = 0; j < 2; ++j) {
                v16bf bf = lds_b_frag(Bs[cur], wn + 16 * j, lane, kb);
#pragma unroll
                for (int i = 0; i < 4; ++i)
                    acc[i][j] = __builtin_amdgcn_wmma_f32_16x16x32_bf16(
                        false, af[i], false, bf, (short)0, acc[i][j], false, false);
            }
        }
        wait_async0();
        __syncthreads();
    }

    const int hi = lane >> 4;
    const int ln = lane & 15;
#pragma unroll
    for (int i = 0; i < 4; ++i) {
#pragma unroll
        for (int j = 0; j < 2; ++j) {
            int n = n0 + wn + 16 * j + ln;
#pragma unroll
            for (int v = 0; v < 8; ++v) {
                int m = m0 + wm + 16 * i + v + 8 * hi;
                out[(size_t)(e * TOK_PER_E + m) * DIM + n] = acc[i][j][v];
            }
        }
    }
}

// ---------------------------------------------------------------------------
extern "C" void kernel_launch(void* const* d_in, const int* in_sizes, int n_in,
                              void* d_out, int out_size, void* d_ws, size_t ws_size,
                              hipStream_t stream) {
    const float* x     = (const float*)d_in[0];   // [16384, 2048]
    const float* probs = (const float*)d_in[1];   // [16384, 1]
    const float* gup   = (const float*)d_in[2];   // [8, 2048, 2816]
    const float* dwn   = (const float*)d_in[3];   // [8, 1408, 2048]
    // d_in[4]: tokens_per_expert (static equal split -> unused)
    float* out = (float*)d_out;

    // workspace layout (bf16 copies, all 16B-aligned)
    char* ws = (char*)d_ws;
    const size_t sz_xb   = (size_t)TOTAL_TOKENS * DIM * 2;               //  64.0 MB
    const size_t sz_gupT = (size_t)N_EXPERTS * 2 * INTER * DIM * 2;      //  92.3 MB
    const size_t sz_dT   = (size_t)N_EXPERTS * DIM * INTER * 2;          //  46.1 MB
    bf16_t* xb   = (bf16_t*)(ws);
    bf16_t* gupT = (bf16_t*)(ws + sz_xb);
    bf16_t* dT   = (bf16_t*)(ws + sz_xb + sz_gupT);
    bf16_t* actb = (bf16_t*)(ws + sz_xb + sz_gupT + sz_dT);

    // 1) bf16 conversion / transposition passes (bandwidth-bound)
    convert_x_kernel<<<(TOTAL_TOKENS * DIM) / 1024, 256, 0, stream>>>(x, xb);
    transpose_gup_kernel<<<dim3(88, 64, N_EXPERTS), dim3(32, 8), 0, stream>>>(gup, gupT);
    transpose_down_kernel<<<dim3(64, 44, N_EXPERTS), dim3(32, 8), 0, stream>>>(dwn, dT);

    // 2) GEMM1 + GEGLU  (grid: 1408/128=11 n-tiles, 2048/128=16 m-tiles, 8 experts)
    moe_gemm1_kernel<<<dim3(11, 16, N_EXPERTS), 256, 0, stream>>>(xb, gupT, probs, actb);

    // 3) GEMM2 -> fp32 out (grid: 2048/128=16 x 16 x 8)
    moe_gemm2_kernel<<<dim3(16, 16, N_EXPERTS), 256, 0, stream>>>(actb, dT, out);
}
